// MSTAGAT_Net_3925600108907
// MI455X (gfx1250) — matrix operand order repaired and checked by reading, not compiled, not run on hardware
//
#include <hip/hip_runtime.h>
#include <hip/hip_bf16.h>
#include <math.h>

typedef __attribute__((ext_vector_type(16))) _Float16 v16h;
typedef __attribute__((ext_vector_type(8)))  _Float16 v8h;
typedef __attribute__((ext_vector_type(8)))  float    v8f;

#define B_     8
#define N_     1600
#define T_     64
#define H_     32
#define HEADS_ 4
#define HD_    8
#define BK_    8
#define FC_    16
#define HOR_   12
#define ROWS_  (B_*N_)     // 12800
#define K_     (FC_*T_)    // 1024

__device__ __forceinline__ float sigm(float v) { return 1.0f / (1.0f + expf(-v)); }

// ---------------------------------------------------------------------------
// Kernel 0: fp_w (32x1024 f32) -> f16 in workspace (B operand of main GEMM)
// ---------------------------------------------------------------------------
__global__ __launch_bounds__(256) void k_prep(const float* __restrict__ fpw,
                                              _Float16* __restrict__ fpw16) {
  int i = blockIdx.x * 256 + threadIdx.x;
  if (i < H_ * K_) fpw16[i] = (_Float16)fpw[i];
}

// ---------------------------------------------------------------------------
// Kernel 1: depthwise conv + BN1 + ReLU -> LDS (f16), then
//           feat(12800x32) = relu(alpha[c]*y1[t]+beta[c]) @ fp_w^T via WMMA.
// 100 blocks x 256 threads (8 waves). Each block: 128 rows. Each wave: 16 rows.
// ---------------------------------------------------------------------------
__global__ __launch_bounds__(256) void k_front_gemm(
    const float* __restrict__ x,
    const float* __restrict__ dw_w, const float* __restrict__ dw_b,
    const float* __restrict__ bn1_g, const float* __restrict__ bn1_b,
    const float* __restrict__ bn1_m, const float* __restrict__ bn1_v,
    const float* __restrict__ pw_w,  const float* __restrict__ pw_b,
    const float* __restrict__ bn2_g, const float* __restrict__ bn2_b,
    const float* __restrict__ bn2_m, const float* __restrict__ bn2_v,
    const _Float16* __restrict__ fpw16, const float* __restrict__ fp_b,
    float* __restrict__ featF)
{
  __shared__ __align__(16) _Float16 y1h[128][72];   // 64 t-values, padded stride
  __shared__ float sAlpha[FC_], sBeta[FC_], sFpb[H_];

  const int tid = threadIdx.x;

  if (tid < FC_) {
    float inv2 = bn2_g[tid] * rsqrtf(bn2_v[tid] + 1e-5f);
    sAlpha[tid] = inv2 * pw_w[tid];
    sBeta[tid]  = (pw_b[tid] - bn2_m[tid]) * inv2 + bn2_b[tid];
  }
  if (tid < H_) sFpb[tid] = fp_b[tid];

  if (tid < 128) {
    const int row = blockIdx.x * 128 + tid;          // global row (b*N + n)
    const int b = row / N_;
    const int n = row - b * N_;
    const float w0 = dw_w[0], w1 = dw_w[1], w2 = dw_w[2], db = dw_b[0];
    const float inv1 = bn1_g[0] * rsqrtf(bn1_v[0] + 1e-5f);
    const float sh1  = bn1_b[0] - bn1_m[0] * inv1;
    const float* xb = x + (size_t)b * T_ * N_ + n;   // x[b, t, n], stride N_ over t
    float xm = 0.0f;
    float xc = xb[0];
    for (int t = 0; t < T_; ++t) {
      float xp = (t < T_ - 1) ? xb[(size_t)(t + 1) * N_] : 0.0f;
      float yv = w0 * xm + w1 * xc + w2 * xp + db;   // conv1d, pad=1
      yv = fmaxf(yv * inv1 + sh1, 0.0f);             // BN1 + ReLU
      y1h[tid][t] = (_Float16)yv;
      xm = xc; xc = xp;
    }
  }
  __syncthreads();

  // ---- WMMA phase: every wave computes a 16x32 tile of feat ----
  const int lane   = tid & 31;
  const int waveId = tid >> 5;
  const int m      = lane & 15;     // row-in-tile (A) / column (B,C)
  const int half   = lane >> 4;     // lane group 0 / 1
  const int lr     = waveId * 16 + m;  // LDS row for the A fragment

  v8f acc0 = {0.f,0.f,0.f,0.f,0.f,0.f,0.f,0.f};
  v8f acc1 = {0.f,0.f,0.f,0.f,0.f,0.f,0.f,0.f};

  for (int kc = 0; kc < 32; ++kc) {          // K = 1024 in chunks of 32
    const int c  = kc >> 1;                  // pointwise channel
    const int t0 = (kc & 1) * 32;            // t-range base
    const float al = sAlpha[c], be = sBeta[c];

    // A fragment (16-bit A 16x32): lane<16 -> K 0..15 ; lane>=16 -> K 16..31
    const int tb = t0 + half * 16;
    v8h g0 = *(const v8h*)&y1h[lr][tb];
    v8h g1 = *(const v8h*)&y1h[lr][tb + 8];
    v16h a;
#pragma unroll
    for (int j = 0; j < 8; ++j) {
      a[j]     = (_Float16)fmaxf(al * (float)g0[j] + be, 0.0f);  // relu(pw+BN2)
      a[8 + j] = (_Float16)fmaxf(al * (float)g1[j] + be, 0.0f);
    }

    // B fragments: lanes 0-15 hold K 0..15, lanes 16-31 hold K 16..31 (contig 32B)
    const int kb = kc * 32 + half * 16;
    v16h b0 = *(const v16h*)(fpw16 + (size_t)m * K_ + kb);
    v16h b1 = *(const v16h*)(fpw16 + (size_t)(16 + m) * K_ + kb);

    acc0 = __builtin_amdgcn_wmma_f32_16x16x32_f16(false, a, false, b0,
                                                  (short)0, acc0, false, false);
    acc1 = __builtin_amdgcn_wmma_f32_16x16x32_f16(false, a, false, b1,
                                                  (short)0, acc1, false, false);
  }

  // C layout: VGPR g -> (M = g + 8*half, N = m). Add bias, store.
  const int rbase = blockIdx.x * 128 + waveId * 16 + half * 8;
  const float bias0 = sFpb[m], bias1 = sFpb[16 + m];
#pragma unroll
  for (int g = 0; g < 8; ++g) {
    const int row = rbase + g;
    featF[(size_t)row * H_ + m]      = acc0[g] + bias0;
    featF[(size_t)row * H_ + 16 + m] = acc1[g] + bias1;
  }
}

// ---------------------------------------------------------------------------
// Kernel 2: qkv chain per row; qe/zinv stored; per-wave kv partial sums.
// 50 blocks x 256 threads, one row per thread. 1600 % 32 == 0 so each wave
// stays inside one batch b -> wave-level shfl reduction is exact.
// ---------------------------------------------------------------------------
__global__ __launch_bounds__(256) void k_qkv(
    const float* __restrict__ featF,
    const float* __restrict__ lo_w, const float* __restrict__ lo_b,
    const float* __restrict__ hi_w, const float* __restrict__ hi_b,
    float* __restrict__ qeF, float* __restrict__ zinvF,
    float* __restrict__ wavePart)
{
  __shared__ float sLoW[24 * 32], sLoB[24], sHiW[96 * 24], sHiB[96];
  const int tid = threadIdx.x;
  for (int i = tid; i < 24 * 32; i += 256) sLoW[i] = lo_w[i];
  for (int i = tid; i < 24;      i += 256) sLoB[i] = lo_b[i];
  for (int i = tid; i < 96 * 24; i += 256) sHiW[i] = hi_w[i];
  for (int i = tid; i < 96;      i += 256) sHiB[i] = hi_b[i];
  __syncthreads();

  const int row = blockIdx.x * 256 + tid;

  float f[H_];
#pragma unroll
  for (int i = 0; i < H_; ++i) f[i] = featF[(size_t)row * H_ + i];

  float t24[24];
  for (int j = 0; j < 24; ++j) {
    float s = sLoB[j];
#pragma unroll
    for (int i = 0; i < H_; ++i) s += f[i] * sLoW[j * H_ + i];
    t24[j] = s;
  }

  float qe[H_], ke[H_], vv[H_];
  for (int j = 0; j < H_; ++j) {
    float sq = sHiB[j], sk = sHiB[32 + j], sv = sHiB[64 + j];
#pragma unroll
    for (int i = 0; i < 24; ++i) {
      sq += t24[i] * sHiW[j * 24 + i];
      sk += t24[i] * sHiW[(32 + j) * 24 + i];
      sv += t24[i] * sHiW[(64 + j) * 24 + i];
    }
    qe[j] = (sq > 0.0f) ? sq + 1.0f : expf(sq);   // elu(q)+1
    ke[j] = (sk > 0.0f) ? sk + 1.0f : expf(sk);   // elu(k)+1
    vv[j] = sv;
    qeF[(size_t)row * H_ + j] = qe[j];
  }

  for (int h = 0; h < HEADS_; ++h) {
    float zs = 1e-8f;
#pragma unroll
    for (int d = 0; d < HD_; ++d) zs += ke[h * HD_ + d];
    zinvF[(size_t)row * HEADS_ + h] = 1.0f / zs;
  }

  // per-wave kv partial: kv[h][d][e] = sum_lanes ke[h,d]*vv[h,e]
  const int lane = tid & 31;
  const int gw   = row >> 5;                       // global wave id, 0..399
  float* wp = wavePart + (size_t)gw * 256;
  for (int h = 0; h < HEADS_; ++h) {
    for (int d = 0; d < HD_; ++d) {
      const float kd = ke[h * HD_ + d];
      for (int e = 0; e < HD_; ++e) {
        float p = kd * vv[h * HD_ + e];
#pragma unroll
        for (int off = 16; off > 0; off >>= 1) p += __shfl_xor(p, off, 32);
        if (lane == 0) wp[h * 64 + d * 8 + e] = p;
      }
    }
  }
}

// ---------------------------------------------------------------------------
// Kernel 3: sum the 50 wave-partials per batch b -> kvAcc[b][h][d][e]
// ---------------------------------------------------------------------------
__global__ __launch_bounds__(256) void k_reduce_kv(const float* __restrict__ wavePart,
                                                   float* __restrict__ kvAcc) {
  const int idx = blockIdx.x * 256 + threadIdx.x;   // 2048 = 8*256
  if (idx >= B_ * HEADS_ * 64) return;
  const int b = idx >> 8, rem = idx & 255;
  float s = 0.0f;
  for (int w = 0; w < 50; ++w) s += wavePart[(size_t)(b * 50 + w) * 256 + rem];
  kvAcc[idx] = s;
}

// ---------------------------------------------------------------------------
// Kernel 4: linear-attention output, out-proj residual, gate MLP, GRU rollout,
// decay blend -> final; writes reg = exp(log_reg)/N (softmax-mean identity).
// ---------------------------------------------------------------------------
__global__ __launch_bounds__(256) void k_tail(
    const float* __restrict__ x, const float* __restrict__ featF,
    const float* __restrict__ qeF, const float* __restrict__ zinvF,
    const float* __restrict__ kvAcc,
    const float* __restrict__ out_lo_w, const float* __restrict__ out_lo_b,
    const float* __restrict__ out_hi_w, const float* __restrict__ out_hi_b,
    const float* __restrict__ hp_w, const float* __restrict__ hp_b,
    const float* __restrict__ gru_wih, const float* __restrict__ gru_whh,
    const float* __restrict__ gru_bih, const float* __restrict__ gru_bhh,
    const float* __restrict__ op_w, const float* __restrict__ op_b,
    const float* __restrict__ log_decay,
    const float* __restrict__ rg1_w, const float* __restrict__ rg1_b,
    const float* __restrict__ rg2_w, const float* __restrict__ rg2_b,
    const float* __restrict__ log_reg,
    float* __restrict__ out)
{
  __shared__ float sOLW[8 * 32], sOLB[8], sOHW[32 * 8], sOHB[32];
  __shared__ float sHpW[8 * 32], sHpB[8], sR1W[8 * 32], sR1B[8];
  __shared__ float sR2W[12 * 8], sR2B[12];
  __shared__ float sWih[24], sWhh[24 * 8], sBih[24], sBhh[24], sOpW[8];

  const int tid = threadIdx.x;
  for (int i = tid; i < 256; i += 256) { sOLW[i] = out_lo_w[i]; sOHW[i] = out_hi_w[i];
                                         sHpW[i] = hp_w[i];     sR1W[i] = rg1_w[i]; }
  for (int i = tid; i < 192; i += 256) sWhh[i] = gru_whh[i];
  for (int i = tid; i < 96;  i += 256) sR2W[i] = rg2_w[i];
  if (tid < 32) sOHB[tid] = out_hi_b[tid];
  if (tid < 24) { sWih[tid] = gru_wih[tid]; sBih[tid] = gru_bih[tid]; sBhh[tid] = gru_bhh[tid]; }
  if (tid < 12) sR2B[tid] = rg2_b[tid];
  if (tid < 8)  { sOLB[tid] = out_lo_b[tid]; sHpB[tid] = hp_b[tid];
                  sR1B[tid] = rg1_b[tid];    sOpW[tid] = op_w[tid]; }
  __syncthreads();

  const int row = blockIdx.x * 256 + tid;
  const int b = row / N_;
  const int n = row - b * N_;

  float feat[H_], qe[H_], att[H_];
#pragma unroll
  for (int i = 0; i < H_; ++i) { feat[i] = featF[(size_t)row * H_ + i];
                                 qe[i]   = qeF[(size_t)row * H_ + i]; }

  const float* kv = kvAcc + (size_t)b * HEADS_ * 64;
  for (int h = 0; h < HEADS_; ++h) {
    const float z = zinvF[(size_t)row * HEADS_ + h];
    for (int e = 0; e < HD_; ++e) {
      float s = 0.0f;
#pragma unroll
      for (int d = 0; d < HD_; ++d) s += qe[h * HD_ + d] * kv[h * 64 + d * 8 + e];
      att[h * HD_ + e] = s * z;
    }
  }

  float o8[8];
  for (int j = 0; j < 8; ++j) {
    float s = sOLB[j];
#pragma unroll
    for (int i = 0; i < H_; ++i) s += att[i] * sOLW[j * H_ + i];
    o8[j] = s;
  }
  for (int j = 0; j < H_; ++j) {
    float s = sOHB[j];
#pragma unroll
    for (int i = 0; i < 8; ++i) s += o8[i] * sOHW[j * 8 + i];
    feat[j] += s;                                    // residual
  }

  float h8[8], g8[8];
  for (int j = 0; j < 8; ++j) {
    float sh = sHpB[j], sg = sR1B[j];
#pragma unroll
    for (int i = 0; i < H_; ++i) { sh += feat[i] * sHpW[j * H_ + i];
                                   sg += feat[i] * sR1W[j * H_ + i]; }
    h8[j] = sh;
    g8[j] = fmaxf(sg, 0.0f);
  }
  float gate[HOR_];
  for (int t = 0; t < HOR_; ++t) {
    float s = sR2B[t];
#pragma unroll
    for (int j = 0; j < 8; ++j) s += g8[j] * sR2W[t * 8 + j];
    gate[t] = sigm(s);
  }

  const float last = x[((size_t)b * T_ + (T_ - 1)) * N_ + n];
  const float opb = op_b[0];
  float cur = last;
  float pred[HOR_];
  for (int st = 0; st < HOR_; ++st) {
    float hn[8];
#pragma unroll
    for (int j = 0; j < 8; ++j) {
      float gh_r = sBhh[j], gh_z = sBhh[8 + j], gh_n = sBhh[16 + j];
#pragma unroll
      for (int i = 0; i < 8; ++i) {
        gh_r += h8[i] * sWhh[j * 8 + i];
        gh_z += h8[i] * sWhh[(8 + j) * 8 + i];
        gh_n += h8[i] * sWhh[(16 + j) * 8 + i];
      }
      const float gx_r = cur * sWih[j]      + sBih[j];
      const float gx_z = cur * sWih[8 + j]  + sBih[8 + j];
      const float gx_n = cur * sWih[16 + j] + sBih[16 + j];
      const float r  = sigm(gx_r + gh_r);
      const float zg = sigm(gx_z + gh_z);
      const float nn = tanhf(gx_n + r * gh_n);
      hn[j] = (1.0f - zg) * nn + zg * h8[j];
    }
    float p = opb;
#pragma unroll
    for (int j = 0; j < 8; ++j) { h8[j] = hn[j]; p += hn[j] * sOpW[j]; }
    pred[st] = p;
    cur = p;
  }

  const float dk = expf(log_decay[0]);
#pragma unroll
  for (int t = 0; t < HOR_; ++t) {
    const float dec = expf(-dk * (float)(t + 1));
    out[(size_t)row * HOR_ + t] = gate[t] * pred[t] + (1.0f - gate[t]) * last * dec;
  }

  // reg = exp(log_reg) * mean(|softmax|) = exp(log_reg) * 1/N  (rows sum to 1)
  if (row == 0) out[(size_t)ROWS_ * HOR_] = expf(log_reg[0]) * (1.0f / (float)N_);
}

// ---------------------------------------------------------------------------
extern "C" void kernel_launch(void* const* d_in, const int* in_sizes, int n_in,
                              void* d_out, int out_size, void* d_ws, size_t ws_size,
                              hipStream_t stream) {
  (void)in_sizes; (void)n_in; (void)out_size; (void)ws_size;
  const float* x        = (const float*)d_in[0];
  const float* dw_w     = (const float*)d_in[1];
  const float* dw_b     = (const float*)d_in[2];
  const float* bn1_g    = (const float*)d_in[3];
  const float* bn1_b    = (const float*)d_in[4];
  const float* bn1_m    = (const float*)d_in[5];
  const float* bn1_v    = (const float*)d_in[6];
  const float* pw_w     = (const float*)d_in[7];
  const float* pw_b     = (const float*)d_in[8];
  const float* bn2_g    = (const float*)d_in[9];
  const float* bn2_b    = (const float*)d_in[10];
  const float* bn2_m    = (const float*)d_in[11];
  const float* bn2_v    = (const float*)d_in[12];
  const float* fp_w     = (const float*)d_in[13];
  const float* fp_b     = (const float*)d_in[14];
  const float* qkv_lo_w = (const float*)d_in[15];
  const float* qkv_lo_b = (const float*)d_in[16];
  const float* qkv_hi_w = (const float*)d_in[17];
  const float* qkv_hi_b = (const float*)d_in[18];
  const float* out_lo_w = (const float*)d_in[19];
  const float* out_lo_b = (const float*)d_in[20];
  const float* out_hi_w = (const float*)d_in[21];
  const float* out_hi_b = (const float*)d_in[22];
  // d_in[23]=u, d_in[24]=v : dead (softmax-mean identity)
  const float* log_reg  = (const float*)d_in[25];
  const float* hp_w     = (const float*)d_in[26];
  const float* hp_b     = (const float*)d_in[27];
  const float* gru_wih  = (const float*)d_in[28];
  const float* gru_whh  = (const float*)d_in[29];
  const float* gru_bih  = (const float*)d_in[30];
  const float* gru_bhh  = (const float*)d_in[31];
  const float* op_w     = (const float*)d_in[32];
  const float* op_b     = (const float*)d_in[33];
  const float* log_decay= (const float*)d_in[34];
  const float* rg1_w    = (const float*)d_in[35];
  const float* rg1_b    = (const float*)d_in[36];
  const float* rg2_w    = (const float*)d_in[37];
  const float* rg2_b    = (const float*)d_in[38];

  char* ws = (char*)d_ws;
  _Float16* fpw16 = (_Float16*)(ws);                                   // 65536 B
  float* featF    = (float*)(ws + 65536);                              // 1,638,400 B
  float* qeF      = (float*)(ws + 65536 + 1638400);                    // 1,638,400 B
  float* zinvF    = (float*)(ws + 65536 + 2 * 1638400);                // 204,800 B
  float* wavePart = (float*)(ws + 65536 + 2 * 1638400 + 204800);       // 409,600 B
  float* kvAcc    = (float*)(ws + 65536 + 2 * 1638400 + 204800 + 409600); // 8,192 B

  float* out = (float*)d_out;

  k_prep<<<128, 256, 0, stream>>>(fp_w, fpw16);

  k_front_gemm<<<100, 256, 0, stream>>>(x, dw_w, dw_b, bn1_g, bn1_b, bn1_m, bn1_v,
                                        pw_w, pw_b, bn2_g, bn2_b, bn2_m, bn2_v,
                                        fpw16, fp_b, featF);

  k_qkv<<<50, 256, 0, stream>>>(featF, qkv_lo_w, qkv_lo_b, qkv_hi_w, qkv_hi_b,
                                qeF, zinvF, wavePart);

  k_reduce_kv<<<8, 256, 0, stream>>>(wavePart, kvAcc);

  k_tail<<<50, 256, 0, stream>>>(x, featF, qeF, zinvF, kvAcc,
                                 out_lo_w, out_lo_b, out_hi_w, out_hi_b,
                                 hp_w, hp_b, gru_wih, gru_whh, gru_bih, gru_bhh,
                                 op_w, op_b, log_decay, rg1_w, rg1_b, rg2_w, rg2_b,
                                 log_reg, out);
}
